// CrossAttention_81535659148074
// MI455X (gfx1250) — matrix operand order repaired and checked by reading, not compiled, not run on hardware
//
#include <hip/hip_runtime.h>

constexpr int CD   = 1024;   // model dim (= K = N of all GEMMs)
constexpr int CNq  = 2048;
constexpr int CNk  = 2048;   // power of two: row>>11
constexpr int CB   = 4;
constexpr int CH   = 16;
constexpr int CHD  = 64;
// SCALE * log2(e): folded into the Q projection so attention logits are
// already in exp2 domain (saves the per-element scale + ln2 muls).
#define QSCALE_LOG2E 0.1803368801111204f   // 0.125 * 1.4426950408889634

typedef __attribute__((ext_vector_type(16))) __bf16 v16bf;
typedef __attribute__((ext_vector_type(8)))  __bf16 v8bf;
typedef __attribute__((ext_vector_type(8)))  float  v8f;
typedef __attribute__((ext_vector_type(4)))  int    v4i;

#if defined(__has_builtin)
#if __has_builtin(__builtin_amdgcn_global_load_async_to_lds_b128)
#define HAS_ASYNC_LDS 1
#endif
#endif

static __device__ __forceinline__ __bf16 f2bf(float f) {
  unsigned u = __builtin_bit_cast(unsigned, f);
  unsigned r = (u + 0x7FFFu + ((u >> 16) & 1u)) >> 16;   // round-to-nearest-even
  unsigned short s = (unsigned short)r;
  __bf16 o;
  __builtin_memcpy(&o, &s, 2);
  return o;
}

static __device__ __forceinline__ v16bf bfcat(v8bf lo, v8bf hi) {
  v16bf r;
#pragma unroll
  for (int i = 0; i < 8; ++i) { r[i] = lo[i]; r[i + 8] = hi[i]; }
  return r;
}

static __device__ __forceinline__ v8f wmma_bf16(v16bf a, v16bf b, v8f c) {
  return __builtin_amdgcn_wmma_f32_16x16x32_bf16(false, a, false, b, (short)0, c,
                                                 false, false);
}

// Copy 16 bytes global -> LDS.  Async path (tracked by ASYNCcnt) when the
// toolchain exposes the builtin; otherwise load + ds_store fallback.
static __device__ __forceinline__ void stage_b128(const __bf16* gp, __bf16* lp) {
#if defined(HAS_ASYNC_LDS)
  void* gnc = (void*)gp;                                       // drop const
  __attribute__((address_space(1))) v4i* g1 =
      (__attribute__((address_space(1))) v4i*)gnc;             // generic -> AS1
  __attribute__((address_space(3))) v4i* l3 =
      (__attribute__((address_space(3))) v4i*)(void*)lp;       // generic -> AS3
  __builtin_amdgcn_global_load_async_to_lds_b128(g1, l3, 0, 0);
#else
  *(v8bf*)lp = *(const v8bf*)gp;
#endif
}

static __device__ __forceinline__ void async_stage_wait() {
#if defined(HAS_ASYNC_LDS)
  asm volatile("s_wait_asynccnt 0" ::: "memory");
#endif
}

// ---------------------------------------------------------------------------
// LayerNorm + cast fp32 -> bf16.  One block (256 threads) per row of D=1024.
// ---------------------------------------------------------------------------
__global__ __launch_bounds__(256) void ca_ln_cast(
    const float* __restrict__ x, const float* __restrict__ g,
    const float* __restrict__ bb, __bf16* __restrict__ out)
{
  const int row = blockIdx.x;
  const float* xr = x + (size_t)row * CD;
  float v[4];
  float s = 0.f, s2 = 0.f;
#pragma unroll
  for (int i = 0; i < 4; ++i) {
    v[i] = xr[threadIdx.x + 256 * i];
    s += v[i]; s2 += v[i] * v[i];
  }
#pragma unroll
  for (int m = 16; m >= 1; m >>= 1) {
    s  += __shfl_xor(s,  m, 32);
    s2 += __shfl_xor(s2, m, 32);
  }
  __shared__ float ws0[8], ws1[8];
  const int lane = threadIdx.x & 31, wid = threadIdx.x >> 5;
  if (lane == 0) { ws0[wid] = s; ws1[wid] = s2; }
  __syncthreads();
  float ts = 0.f, ts2 = 0.f;
#pragma unroll
  for (int i = 0; i < 8; ++i) { ts += ws0[i]; ts2 += ws1[i]; }
  const float mean = ts * (1.0f / CD);
  const float var  = ts2 * (1.0f / CD) - mean * mean;
  const float rstd = rsqrtf(var + 1e-5f);
  __bf16* orow = out + (size_t)row * CD;
#pragma unroll
  for (int i = 0; i < 4; ++i) {
    const int c = threadIdx.x + 256 * i;
    orow[c] = f2bf((v[i] - mean) * rstd * g[c] + bb[c]);
  }
}

// ---------------------------------------------------------------------------
// Cast + transpose weight W[K,N] fp32 -> Wt[N,K] bf16 (32x32 LDS tiles).
// ---------------------------------------------------------------------------
__global__ __launch_bounds__(256) void ca_wtrans(
    const float* __restrict__ W, __bf16* __restrict__ Wt)
{
  __shared__ float tile[32][33];
  const int n0 = blockIdx.x * 32, k0 = blockIdx.y * 32;
  const int tx = threadIdx.x, ty = threadIdx.y;   // 32 x 8
#pragma unroll
  for (int j = 0; j < 4; ++j)
    tile[ty + 8 * j][tx] = W[(size_t)(k0 + ty + 8 * j) * CD + n0 + tx];
  __syncthreads();
#pragma unroll
  for (int j = 0; j < 4; ++j)
    Wt[(size_t)(n0 + ty + 8 * j) * CD + k0 + tx] = f2bf(tile[tx][ty + 8 * j]);
}

// ---------------------------------------------------------------------------
// GEMM: out[M,1024] = A[M,1024](bf16) @ Wt[1024,1024]^T(bf16) + bias.
// Wave tile 32x64 (2x4 wmma accumulators).  Block = 4 waves -> 128x64 tile.
// The shared 64-col B tile is double-buffer staged into LDS (async when
// available), giving 4x reuse across the block's waves.
// MODE 0: bf16 row-major; MODE 1: bf16 transposed into Vt[B, 1024, Nk];
// MODE 2: f32 row-major (final output);
// MODE 3: bf16 row-major scaled by SCALE*log2e (Q projection).
// ---------------------------------------------------------------------------
template <int MODE>
__global__ __launch_bounds__(128) void ca_gemm(
    const __bf16* __restrict__ A, const __bf16* __restrict__ Wt,
    const float* __restrict__ bias, void* __restrict__ outp)
{
  __shared__ __bf16 Bl[2][64][32];   // double-buffered B tile, 8 KB
  const int tid  = threadIdx.x;
  const int lane = tid & 31;
  const int wid  = tid >> 5;
  const int hl   = lane & 15;
  const int hi16 = (lane >> 4) & 1;
  const int mbase = (blockIdx.y * 4 + wid) * 32;
  const int nbase = blockIdx.x * 64;

  // Staging assignment: thread t copies 32B of row (t>>1), half (t&1).
  const int srow = tid >> 1;
  const int scol = (tid & 1) * 16;
  const __bf16* sg = Wt + (size_t)(nbase + srow) * CD + scol;

  v8f acc[2][4] = {};
  const __bf16* ar0 = A + (size_t)(mbase + hl) * CD;
  const __bf16* ar1 = A + (size_t)(mbase + 16 + hl) * CD;

  // Prologue: stage k0 = 0 into buffer 0.
  stage_b128(sg,     &Bl[0][srow][scol]);
  stage_b128(sg + 8, &Bl[0][srow][scol] + 8);

  for (int k0 = 0; k0 < CD; k0 += 32) {
    const int cur = (k0 >> 5) & 1;
    async_stage_wait();      // own async chunks landed
    __syncthreads();         // everyone's chunks landed; prev buffer free
    if (k0 + 32 < CD) {
      const __bf16* sgn = sg + k0 + 32;
      __bf16* ld = &Bl[cur ^ 1][srow][scol];
      stage_b128(sgn,     ld);
      stage_b128(sgn + 8, ld + 8);
    }

    v16bf afr[2];
    {
      const __bf16* p0 = ar0 + k0 + hi16 * 8;
      afr[0] = bfcat(*(const v8bf*)p0, *(const v8bf*)(p0 + 16));
      const __bf16* p1 = ar1 + k0 + hi16 * 8;
      afr[1] = bfcat(*(const v8bf*)p1, *(const v8bf*)(p1 + 16));
    }
    v16bf bfr[4];
#pragma unroll
    for (int jn = 0; jn < 4; ++jn) {
      const __bf16* bp = &Bl[cur][16 * jn + hl][hi16 * 8];
      bfr[jn] = bfcat(*(const v8bf*)bp, *(const v8bf*)(bp + 16));
    }
#pragma unroll
    for (int im = 0; im < 2; ++im)
#pragma unroll
      for (int jn = 0; jn < 4; ++jn)
        acc[im][jn] = wmma_bf16(afr[im], bfr[jn], acc[im][jn]);
  }

#pragma unroll
  for (int jn = 0; jn < 4; ++jn) {
    const int col = nbase + 16 * jn + hl;
    const float bv = bias[col];
#pragma unroll
    for (int im = 0; im < 2; ++im) {
#pragma unroll
      for (int r = 0; r < 8; ++r) {
        const int row = mbase + im * 16 + r + hi16 * 8;
        const float val = acc[im][jn][r] + bv;
        if constexpr (MODE == 2) {
          ((float*)outp)[(size_t)row * CD + col] = val;
        } else if constexpr (MODE == 0) {
          ((__bf16*)outp)[(size_t)row * CD + col] = f2bf(val);
        } else if constexpr (MODE == 3) {
          ((__bf16*)outp)[(size_t)row * CD + col] = f2bf(val * QSCALE_LOG2E);
        } else {
          const int bidx = row >> 11;          // / CNk
          const int kk   = row & (CNk - 1);
          ((__bf16*)outp)[((size_t)bidx * CD + col) * CNk + kk] = f2bf(val);
        }
      }
    }
  }
}

// ---------------------------------------------------------------------------
// Flash-attention style.  One wave per 16-query tile, 4 waves/block.
// Q is pre-scaled by SCALE*log2e, so S accumulators are exp2-domain logits:
// softmax uses bare v_exp_f32 (exp2f) with no extra multiplies.
// Q[B,Nq,D] bf16 (A frags), K[B,Nk,D] bf16 (B frags: d contiguous per key),
// Vt[B,D,Nk] bf16 (B frags: k contiguous per d-col).  O[B,Nq,D] bf16.
// ---------------------------------------------------------------------------
__global__ __launch_bounds__(128) void ca_attn(
    const __bf16* __restrict__ Q, const __bf16* __restrict__ K,
    const __bf16* __restrict__ Vt, __bf16* __restrict__ O)
{
  __shared__ __bf16 Pl[4][16][32];
  const int lane = threadIdx.x & 31;
  const int wid  = threadIdx.x >> 5;
  const int hl   = lane & 15;
  const int hi16 = (lane >> 4) & 1;
  const int h = blockIdx.y;
  const int b = blockIdx.z;
  const int q0 = (blockIdx.x * 4 + wid) * 16;

  // Q tile A-fragments, resident for the whole key loop.
  const __bf16* qrow = Q + ((size_t)b * CNq + q0 + hl) * CD + h * CHD;
  v16bf aQ[2];
#pragma unroll
  for (int f = 0; f < 2; ++f) {
    const __bf16* p = qrow + 32 * f + hi16 * 8;
    aQ[f] = bfcat(*(const v8bf*)p, *(const v8bf*)(p + 16));
  }

  float mrow[8], lrow[8];
  v8f accO[4] = {};
#pragma unroll
  for (int r = 0; r < 8; ++r) { mrow[r] = -3.0e38f; lrow[r] = 0.f; }

  const __bf16* kbase = K + (size_t)b * CNk * CD + h * CHD;
  const __bf16* vbase = Vt + ((size_t)b * CD + h * CHD) * (size_t)CNk;

  for (int kb = 0; kb < CNk; kb += 32) {
    if (kb + 32 < CNk) {   // pull next K/V tiles toward the WGP
      __builtin_prefetch(kbase + (size_t)(kb + 32 + hl) * CD, 0, 3);
      __builtin_prefetch(vbase + (size_t)hl * CNk + kb + 32, 0, 3);
    }

    // S = Q K^T for two 16-key subtiles (accumulate over d = 64).
    v8f sv[2];
#pragma unroll
    for (int sub = 0; sub < 2; ++sub) {
      const __bf16* kp = kbase + (size_t)(kb + 16 * sub + hl) * CD + hi16 * 8;
      v16bf b0 = bfcat(*(const v8bf*)kp,        *(const v8bf*)(kp + 16));
      v16bf b1 = bfcat(*(const v8bf*)(kp + 32), *(const v8bf*)(kp + 48));
      v8f t = {};
      t = wmma_bf16(aQ[0], b0, t);
      t = wmma_bf16(aQ[1], b1, t);
      sv[sub] = t;
    }

    // Online softmax (exp2 domain) across 32 keys.
    float newm[8], alpha[8], psum[8];
#pragma unroll
    for (int r = 0; r < 8; ++r) {
      const float s0 = sv[0][r];
      const float s1 = sv[1][r];
      float rm = fmaxf(s0, s1);
#pragma unroll
      for (int msk = 8; msk >= 1; msk >>= 1)
        rm = fmaxf(rm, __shfl_xor(rm, msk, 32));
      const float nm = fmaxf(mrow[r], rm);
      newm[r]  = nm;
      alpha[r] = exp2f(mrow[r] - nm);
      const float p0 = exp2f(s0 - nm);
      const float p1 = exp2f(s1 - nm);
      Pl[wid][r + hi16 * 8][hl]      = f2bf(p0);
      Pl[wid][r + hi16 * 8][16 + hl] = f2bf(p1);
      float ps = p0 + p1;
#pragma unroll
      for (int msk = 8; msk >= 1; msk >>= 1)
        ps += __shfl_xor(ps, msk, 32);
      psum[r] = ps;
    }
#pragma unroll
    for (int r = 0; r < 8; ++r) {
      lrow[r] = lrow[r] * alpha[r] + psum[r];
      mrow[r] = newm[r];
    }
#pragma unroll
    for (int jd = 0; jd < 4; ++jd)
#pragma unroll
      for (int r = 0; r < 8; ++r) accO[jd][r] *= alpha[r];

    // All ds_stores above must land before cross-lane ds_loads below.
    asm volatile("s_wait_dscnt 0" ::: "memory");

    // Re-load P in A-fragment layout.
    const __bf16* pp = &Pl[wid][hl][0];
    v16bf aP = bfcat(*(const v8bf*)(pp + hi16 * 8),
                     *(const v8bf*)(pp + 16 + hi16 * 8));

    // O += P (16x32) @ V (32x64)
#pragma unroll
    for (int jd = 0; jd < 4; ++jd) {
      const __bf16* vp = vbase + (size_t)(16 * jd + hl) * CNk + kb + hi16 * 8;
      v16bf bV = bfcat(*(const v8bf*)vp, *(const v8bf*)(vp + 16));
      accO[jd] = wmma_bf16(aP, bV, accO[jd]);
    }
  }

  // Normalize and store O as bf16.
  __bf16* obase = O + ((size_t)b * CNq + q0) * CD + h * CHD;
#pragma unroll
  for (int r = 0; r < 8; ++r) {
    const float inv = 1.0f / lrow[r];
    const int row = r + hi16 * 8;
#pragma unroll
    for (int jd = 0; jd < 4; ++jd)
      obase[(size_t)row * CD + 16 * jd + hl] = f2bf(accO[jd][r] * inv);
  }
}

// ---------------------------------------------------------------------------
extern "C" void kernel_launch(void* const* d_in, const int* in_sizes, int n_in,
                              void* d_out, int out_size, void* d_ws, size_t ws_size,
                              hipStream_t stream) {
  (void)in_sizes; (void)n_in; (void)out_size; (void)ws_size;
  const float* query   = (const float*)d_in[0];
  const float* context = (const float*)d_in[1];
  const float* Wq = (const float*)d_in[2];  const float* bq = (const float*)d_in[3];
  const float* Wk = (const float*)d_in[4];  const float* bk = (const float*)d_in[5];
  const float* Wv = (const float*)d_in[6];  const float* bv = (const float*)d_in[7];
  const float* Wo = (const float*)d_in[8];  const float* bo = (const float*)d_in[9];
  const float* gq    = (const float*)d_in[10]; const float* betq  = (const float*)d_in[11];
  const float* gkv   = (const float*)d_in[12]; const float* betkv = (const float*)d_in[13];

  const int Mq = CB * CNq, Mk = CB * CNk;   // 8192 each

  char* ws = (char*)d_ws;
  size_t off = 0;
  auto take = [&](size_t bytes) -> char* {
    char* p = ws + off;
    off += (bytes + 255) & ~(size_t)255;
    return p;
  };
  __bf16* qln = (__bf16*)take((size_t)Mq * CD * 2);
  __bf16* cln = (__bf16*)take((size_t)Mk * CD * 2);
  __bf16* WqT = (__bf16*)take((size_t)CD * CD * 2);
  __bf16* WkT = (__bf16*)take((size_t)CD * CD * 2);
  __bf16* WvT = (__bf16*)take((size_t)CD * CD * 2);
  __bf16* WoT = (__bf16*)take((size_t)CD * CD * 2);
  __bf16* Qp  = (__bf16*)take((size_t)Mq * CD * 2);
  __bf16* Kp  = (__bf16*)take((size_t)Mk * CD * 2);
  __bf16* Vt  = (__bf16*)take((size_t)CB * CD * CNk * 2);
  __bf16* Ob  = (__bf16*)take((size_t)Mq * CD * 2);

  // LayerNorm + cast.
  ca_ln_cast<<<Mq, 256, 0, stream>>>(query,   gq,  betq,  qln);
  ca_ln_cast<<<Mk, 256, 0, stream>>>(context, gkv, betkv, cln);

  // Weight transpose + cast.
  dim3 tb(32, 8), tg(CD / 32, CD / 32);
  ca_wtrans<<<tg, tb, 0, stream>>>(Wq, WqT);
  ca_wtrans<<<tg, tb, 0, stream>>>(Wk, WkT);
  ca_wtrans<<<tg, tb, 0, stream>>>(Wv, WvT);
  ca_wtrans<<<tg, tb, 0, stream>>>(Wo, WoT);

  // Projections (grid: N/64 x M/128).  Q is pre-scaled (MODE 3).
  dim3 gg(CD / 64, Mq / 128);
  ca_gemm<3><<<gg, 128, 0, stream>>>(qln, WqT, bq, Qp);
  ca_gemm<0><<<gg, 128, 0, stream>>>(cln, WkT, bk, Kp);
  ca_gemm<1><<<gg, 128, 0, stream>>>(cln, WvT, bv, Vt);

  // Attention.
  ca_attn<<<dim3(CNq / 64, CH, CB), 128, 0, stream>>>(Qp, Kp, Vt, Ob);

  // Output projection (fp32 into d_out).
  ca_gemm<2><<<gg, 128, 0, stream>>>(Ob, WoT, bo, d_out);
}
